// Attention_15934328668397
// MI455X (gfx1250) — matrix operand order repaired and checked by reading, not compiled, not run on hardware
//
#include <hip/hip_runtime.h>

#define B_DIM 2
#define L_DIM 2048
#define D_DIM 2048
#define NH 16
#define KHD 8
#define HD 128
#define BL (B_DIM * L_DIM)
#define EPS_F 1e-6f
#define K_MASK_F (-2.3819763e+38f)

typedef __attribute__((ext_vector_type(16))) __bf16 v16bf;
typedef __attribute__((ext_vector_type(8)))  float v8f;
typedef __attribute__((ext_vector_type(8)))  unsigned int v8u;
typedef __attribute__((ext_vector_type(4)))  unsigned int tdm_u4;
typedef __attribute__((ext_vector_type(8)))  int tdm_i8;
typedef __attribute__((ext_vector_type(4)))  int tdm_i4;

__device__ __forceinline__ unsigned short f32_to_bf16(float f) {
  union { float f; unsigned int u; } x; x.f = f;
  unsigned int r = x.u + 0x7FFFu + ((x.u >> 16) & 1u);
  return (unsigned short)(r >> 16);
}
__device__ __forceinline__ unsigned int pack_bf16x2(float lo, float hi) {
  return (unsigned int)f32_to_bf16(lo) | ((unsigned int)f32_to_bf16(hi) << 16);
}
__device__ __forceinline__ v8u make_frag(uint4 a, uint4 b) {
  v8u r; r[0]=a.x; r[1]=a.y; r[2]=a.z; r[3]=a.w; r[4]=b.x; r[5]=b.y; r[6]=b.z; r[7]=b.w;
  return r;
}
// A-matrix 16x32 bf16 fragment: lane-low half holds K 0..7 & 16..23, high half 8..15 & 24..31
__device__ __forceinline__ v8u ldA(const unsigned short* rowPtr, int lh) {
  const unsigned short* p = rowPtr + lh * 8;
  return make_frag(*(const uint4*)p, *(const uint4*)(p + 16));
}
// B-matrix 32x16 bf16 fragment: lane-low half holds K 0..7 & 16..23, high half 8..15 & 24..31
__device__ __forceinline__ v8u ldB(const unsigned short* colPtr, int lh) {
  const unsigned short* p = colPtr + lh * 8;
  return make_frag(*(const uint4*)p, *(const uint4*)(p + 16));
}
__device__ __forceinline__ v8f wmma_bf16(v8u a, v8u b, v8f c) {
  return __builtin_amdgcn_wmma_f32_16x16x32_bf16(
      false, __builtin_bit_cast(v16bf, a),
      false, __builtin_bit_cast(v16bf, b),
      (short)0, c, false, false);
}

// ---------------------------------------------------------------------------
// Tensor Data Mover: 2-D tile (2-byte elements) global -> LDS, with LDS row
// padding applied by the DMA. D# per CDNA5 ISA.
// pad codes: interval c -> 2^(c+1) DWORDs between pads; amount c -> (c+1) DWORDs.
// Toolchain uses the 6-arg builtin (extra zero-filled i32x8 before cpol).
// ---------------------------------------------------------------------------
__device__ __forceinline__ void tdm_load_2d(unsigned lds_byte_off, const void* gp,
                                            unsigned tensor_d0, unsigned tensor_d1,
                                            unsigned tile_d0, unsigned tile_d1,
                                            unsigned d0_stride,
                                            unsigned pad_interval_c, unsigned pad_amount_c) {
  unsigned long long ga = (unsigned long long)(size_t)gp;
  tdm_u4 g0;
  g0[0] = 1u;                                            // count=1 (user descriptor)
  g0[1] = lds_byte_off;                                  // lds_addr
  g0[2] = (unsigned)(ga & 0xFFFFFFFFu);                  // global_addr[31:0]
  g0[3] = (unsigned)((ga >> 32) & 0x01FFFFFFu) | 0x80000000u;  // addr[56:32] | type=2
  tdm_i8 g1;
  g1[0] = (int)((1u << 16) | (1u << 20) |                // data_size=2B, pad_enable
                (pad_interval_c << 22) | (pad_amount_c << 25));
  g1[1] = (int)((tensor_d0 & 0xFFFFu) << 16);            // tensor_dim0 lo16 @ [63:48]
  g1[2] = (int)(((tensor_d0 >> 16) & 0xFFFFu) | ((tensor_d1 & 0xFFFFu) << 16));
  g1[3] = (int)(((tensor_d1 >> 16) & 0xFFFFu) | (tile_d0 << 16));
  g1[4] = (int)tile_d1;                                  // tile_dim1 | tile_dim2=0
  g1[5] = (int)d0_stride;                                // tensor_dim0_stride[31:0]
  g1[6] = 0;
  g1[7] = 0;
  tdm_i4 g2; g2[0] = 1; g2[1] = 1; g2[2] = 0; g2[3] = 0; // tensor_dim2/3 = 1
  tdm_i4 g3; g3[0] = 0; g3[1] = (int)(1u << 16); g3[2] = 0; g3[3] = 0; // tensor_dim4 = 1
  tdm_i8 gx = {};                                        // extra operand (zero-filled)
  __builtin_amdgcn_tensor_load_to_lds(g0, g1, g2, g3, gx, 0);
}

// ---------------------------------------------------------------------------
// GEMM: C[M x Ncols] = A[M x kdim] * B[kdim x Ncols], bf16 WMMA, f32 accum.
// Block tile 128x64, 256 thr = 8 waves (4x2), wave tile 32x32 (2x2 WMMA).
// A_BF16: A already bf16 in global -> staged by TDM (wave 0 issues).
// ---------------------------------------------------------------------------
#define GM_LDS 40  // padded LDS stride in bf16 elements (conflict-free b128)

template <int A_BF16>
__global__ __launch_bounds__(256) void gemm_wmma(
    const void* __restrict__ Ap, int lda, const float* __restrict__ Bw, int ldb,
    float* __restrict__ C, int ldc, int kdim) {
  __shared__ __align__(16) unsigned short As[128 * GM_LDS];
  __shared__ __align__(16) unsigned short Bs[64 * GM_LDS];
  const int t = threadIdx.x;
  const int lane = t & 31, wave = t >> 5;
  const int wM = wave & 3, wN = wave >> 2;
  const int lh = lane >> 4, ll = lane & 15;
  const int mBase = blockIdx.y * 128;
  const int nBase = blockIdx.x * 64;

  const float* Af = (const float*)Ap;
  const unsigned short* Ab = (const unsigned short*)Ap;

  v8f zero = {};
  v8f acc[2][2];
  acc[0][0] = zero; acc[0][1] = zero; acc[1][0] = zero; acc[1][1] = zero;

  for (int k0 = 0; k0 < kdim; k0 += 32) {
    if (A_BF16) {
      // A tile (128 rows x 32 bf16) via Tensor Data Mover, pad 16DW->+4DW (stride 40)
      if (wave == 0)
        tdm_load_2d((unsigned)(size_t)(void*)As,
                    Ab + (size_t)mBase * lda + k0,
                    (unsigned)lda, 4096u, 32u, 128u, (unsigned)lda, 3u, 3u);
    } else {
      // stage A tile (128 x 32) as bf16 pairs (f32 -> bf16 conversion)
#pragma unroll
      for (int i = 0; i < 8; i++) {
        int p = t + i * 256;          // pair index, 2048 pairs
        int row = p >> 4, kp = p & 15;
        const float* s = &Af[(size_t)(mBase + row) * lda + k0 + kp * 2];
        *(unsigned int*)&As[row * GM_LDS + kp * 2] = pack_bf16x2(s[0], s[1]);
      }
    }
    // stage B tile (32 x 64) column-major as bf16
#pragma unroll
    for (int i = 0; i < 8; i++) {
      int e = t + i * 256;          // 2048 elements
      int col = e & 63, kk = e >> 6;
      Bs[col * GM_LDS + kk] = f32_to_bf16(Bw[(size_t)(k0 + kk) * ldb + nBase + col]);
    }
    if (k0 + 32 < kdim) {           // prefetch next B tile rows
      __builtin_prefetch(&Bw[(size_t)(k0 + 32 + (t >> 6)) * ldb + nBase + (t & 63)], 0, 1);
    }
    __builtin_amdgcn_s_wait_tensorcnt(0);
    __syncthreads();
    v8u af[2], bf[2];
#pragma unroll
    for (int i = 0; i < 2; i++)
      af[i] = ldA(&As[(wM * 32 + i * 16 + ll) * GM_LDS], lh);
#pragma unroll
    for (int j = 0; j < 2; j++)
      bf[j] = ldB(&Bs[(wN * 32 + j * 16 + ll) * GM_LDS], lh);
#pragma unroll
    for (int i = 0; i < 2; i++)
#pragma unroll
      for (int j = 0; j < 2; j++)
        acc[i][j] = wmma_bf16(af[i], bf[j], acc[i][j]);
    __syncthreads();
  }
#pragma unroll
  for (int i = 0; i < 2; i++)
#pragma unroll
    for (int j = 0; j < 2; j++)
#pragma unroll
      for (int r = 0; r < 8; r++) {
        int row = mBase + wM * 32 + i * 16 + r + 8 * lh;
        int col = nBase + wN * 32 + j * 16 + ll;
        C[(size_t)row * ldc + col] = acc[i][j][r];
      }
}

// ---------------------------------------------------------------------------
// RMS-norm + RoPE, one wave per (b,l,head) row of 128. mode<2: norm+rope,
// output [b][head][l][h] bf16. mode==2 (V): bf16 convert, output TRANSPOSED
// [b][head][h][l] so flash-attn V tiles are contiguous-row TDM loads.
// ---------------------------------------------------------------------------
__global__ __launch_bounds__(256) void normrope_kernel(
    const float* __restrict__ qkv, int colOff, int nheads,
    const float* __restrict__ normw, const int* __restrict__ pos_ids,
    unsigned short* __restrict__ outp, int mode) {
  const int wave = threadIdx.x >> 5, lane = threadIdx.x & 31;
  const long rowId = (long)blockIdx.x * 8 + wave;
  const int n = (int)(rowId % nheads);
  const long bl = rowId / nheads;
  const int b = (int)(bl >> 11), l = (int)(bl & 2047);
  const float* src = qkv + bl * 4096 + colOff + n * HD;

  float v[4];
#pragma unroll
  for (int i = 0; i < 4; i++) v[i] = src[lane + 32 * i];
  if (mode == 2) {
    unsigned short* dstT = outp + (size_t)(b * nheads + n) * HD * L_DIM + l;
#pragma unroll
    for (int i = 0; i < 4; i++) dstT[(size_t)(lane + 32 * i) * L_DIM] = f32_to_bf16(v[i]);
    return;
  }
  unsigned short o[4];
  float ss = v[0]*v[0] + v[1]*v[1] + v[2]*v[2] + v[3]*v[3];
#pragma unroll
  for (int off = 1; off < 32; off <<= 1) ss += __shfl_xor(ss, off, 32);
  float inv = 1.0f / sqrtf(ss * (1.0f / 128.0f) + EPS_F);
#pragma unroll
  for (int i = 0; i < 4; i++) v[i] = v[i] * normw[lane + 32 * i] * inv;
  float pos = (float)pos_ids[b * L_DIM + l];
#pragma unroll
  for (int i = 0; i < 2; i++) {
    float hidx = (float)(lane + 32 * i);
    // timescale = theta^(h/64); ln(1e6)=13.815510557964274
    float ang = pos * expf(-hidx * (13.815510557964274f / 64.0f));
    float sn, cs; sincosf(ang, &sn, &cs);
    float fst = v[i], sec = v[i + 2];
    o[i]     = f32_to_bf16(fst * cs - sec * sn);
    o[i + 2] = f32_to_bf16(sec * cs + fst * sn);
  }
  unsigned short* dst = outp + ((size_t)(b * nheads + n) * L_DIM + l) * HD;
#pragma unroll
  for (int i = 0; i < 4; i++) dst[lane + 32 * i] = o[i];
}

// ---------------------------------------------------------------------------
// Flash attention: 4 waves, 64 q-rows per block, 64-wide s tiles, causal,
// online softmax; QK^T and PV on WMMA bf16. K/V tiles staged by TDM.
// GQA: kv head = n>>1. V is pre-transposed [b][kvh][h][l].
// ---------------------------------------------------------------------------
#define KS_LD 136  // 128 + 8 pad (bf16)
#define VS_LD 72   // 64 + 8 pad
#define PS_LD 72

__global__ __launch_bounds__(128) void flash_attn(
    const unsigned short* __restrict__ qb, const unsigned short* __restrict__ kb,
    const unsigned short* __restrict__ vbT, unsigned short* __restrict__ aout) {
  __shared__ __align__(16) unsigned short Ks[64 * KS_LD];   // [s][h]
  __shared__ __align__(16) unsigned short Vs[HD * VS_LD];   // [h][s]
  __shared__ __align__(16) unsigned short Ps[4 * 16 * PS_LD];
  const int t = threadIdx.x;
  const int lane = t & 31, wave = t >> 5;
  const int lh = lane >> 4, ll = lane & 15;
  const int qTile = blockIdx.x, n = blockIdx.y, b = blockIdx.z;
  const int kvh = n >> 1;
  const float scale = 0.08838834764831845f;  // 1/sqrt(128)

  // Q fragments held in registers for the whole block (A-frag pattern).
  const unsigned short* qrow =
      qb + ((size_t)(b * NH + n) * L_DIM + qTile * 64 + wave * 16 + ll) * HD;
  v8u qf[4];
#pragma unroll
  for (int f = 0; f < 4; f++) qf[f] = ldA(qrow + f * 32, lh);

  v8f zero = {};
  float mrow[8], lrow[8];
  v8f oacc[8];
#pragma unroll
  for (int r = 0; r < 8; r++) { mrow[r] = -3.4e38f; lrow[r] = 0.0f; }
#pragma unroll
  for (int c = 0; c < 8; c++) oacc[c] = zero;

  const unsigned short* kbase  = kb  + (size_t)(b * KHD + kvh) * L_DIM * HD;
  const unsigned short* vbaseT = vbT + (size_t)(b * KHD + kvh) * HD * L_DIM;
  unsigned short* pw = &Ps[wave * 16 * PS_LD];

  for (int sT = 0; sT <= qTile; sT++) {
    // K tile: 64 rows x 128 bf16, pad 64DW->+4DW (stride 136). V tile: 128 rows
    // x 64 bf16, pad 32DW->+4DW (stride 72). One TDM issue per tile.
    if (wave == 0)
      tdm_load_2d((unsigned)(size_t)(void*)Ks, kbase + (size_t)sT * 64 * HD,
                  (unsigned)HD, (unsigned)L_DIM, (unsigned)HD, 64u, (unsigned)HD, 5u, 3u);
    else if (wave == 1)
      tdm_load_2d((unsigned)(size_t)(void*)Vs, vbaseT + (size_t)sT * 64,
                  (unsigned)L_DIM, (unsigned)HD, 64u, (unsigned)HD, (unsigned)L_DIM, 4u, 3u);
    __builtin_amdgcn_s_wait_tensorcnt(0);
    __syncthreads();

    // scores S = Q K^T  (4 Kdim steps x 4 col tiles)
    v8f sc[4];
#pragma unroll
    for (int c = 0; c < 4; c++) sc[c] = zero;
#pragma unroll
    for (int f = 0; f < 4; f++)
#pragma unroll
      for (int c = 0; c < 4; c++)
        sc[c] = wmma_bf16(qf[f], ldB(&Ks[(c * 16 + ll) * KS_LD + f * 32], lh), sc[c]);

    // scale + causal mask + row max
    const int qi0 = qTile * 64 + wave * 16;
    float nm[8];
#pragma unroll
    for (int r = 0; r < 8; r++) nm[r] = -3.4e38f;
#pragma unroll
    for (int c = 0; c < 4; c++) {
      int sIdx = sT * 64 + c * 16 + ll;
#pragma unroll
      for (int r = 0; r < 8; r++) {
        int qIdx = qi0 + r + 8 * lh;
        float sv = sc[c][r] * scale;
        if (sIdx > qIdx) sv = K_MASK_F;
        sc[c][r] = sv;
        nm[r] = fmaxf(nm[r], sv);
      }
    }
#pragma unroll
    for (int r = 0; r < 8; r++) {
      float m = nm[r];
#pragma unroll
      for (int off = 1; off < 16; off <<= 1) m = fmaxf(m, __shfl_xor(m, off, 32));
      nm[r] = fmaxf(m, mrow[r]);
    }
    // probabilities -> LDS (bf16), row sums, rescale accumulators
    float rs[8];
#pragma unroll
    for (int r = 0; r < 8; r++) rs[r] = 0.0f;
#pragma unroll
    for (int c = 0; c < 4; c++)
#pragma unroll
      for (int r = 0; r < 8; r++) {
        float p = expf(sc[c][r] - nm[r]);
        rs[r] += p;
        pw[(r + 8 * lh) * PS_LD + c * 16 + ll] = f32_to_bf16(p);
      }
#pragma unroll
    for (int r = 0; r < 8; r++) {
      float s = rs[r];
#pragma unroll
      for (int off = 1; off < 16; off <<= 1) s += __shfl_xor(s, off, 32);
      float alpha = expf(mrow[r] - nm[r]);
      lrow[r] = lrow[r] * alpha + s;
      mrow[r] = nm[r];
#pragma unroll
      for (int c = 0; c < 8; c++) oacc[c][r] *= alpha;
    }
    asm volatile("s_wait_dscnt 0" ::: "memory");  // same-wave LDS RAW on Ps

    // O += P @ V  (2 Kdim steps x 8 col tiles)
#pragma unroll
    for (int f = 0; f < 2; f++) {
      v8u afr = ldA(&pw[ll * PS_LD + f * 32], lh);
#pragma unroll
      for (int c = 0; c < 8; c++)
        oacc[c] = wmma_bf16(afr, ldB(&Vs[(c * 16 + ll) * VS_LD + f * 32], lh), oacc[c]);
    }
    __syncthreads();
  }
  // epilogue: normalize, write bf16 [bl][n*128+h]
#pragma unroll
  for (int r = 0; r < 8; r++) {
    float inv = 1.0f / lrow[r];
    int qr = qTile * 64 + wave * 16 + r + 8 * lh;
    unsigned short* dst = aout + ((size_t)(b * L_DIM) + qr) * (NH * HD) + n * HD;
#pragma unroll
    for (int c = 0; c < 8; c++) dst[c * 16 + ll] = f32_to_bf16(oacc[c][r] * inv);
  }
}

// ---------------------------------------------------------------------------
extern "C" void kernel_launch(void* const* d_in, const int* in_sizes, int n_in,
                              void* d_out, int out_size, void* d_ws, size_t ws_size,
                              hipStream_t stream) {
  (void)in_sizes; (void)n_in; (void)out_size; (void)ws_size;
  const float* x   = (const float*)d_in[0];
  const int*   pos = (const int*)d_in[1];
  // d_in[2] = attn_mask (tril) -> computed causally in-kernel
  const float* wq  = (const float*)d_in[3];
  const float* wk  = (const float*)d_in[4];
  const float* wv  = (const float*)d_in[5];
  const float* wo  = (const float*)d_in[6];
  const float* qnw = (const float*)d_in[7];
  const float* knw = (const float*)d_in[8];
  float* out = (float*)d_out;

  char* ws = (char*)d_ws;
  float* qkv = (float*)ws;                                                // 64 MB f32
  unsigned short* qb = (unsigned short*)(ws + (size_t)64 * 1024 * 1024);  // 16 MB
  unsigned short* kb = (unsigned short*)(ws + (size_t)80 * 1024 * 1024);  //  8 MB
  unsigned short* vb = (unsigned short*)(ws + (size_t)88 * 1024 * 1024);  //  8 MB (transposed)
  unsigned short* ao = (unsigned short*)ws;  // aliases qkv (dead after normrope)

  dim3 blk(256);
  // QKV projections: X(4096x2048) * W -> qkv columns [0,2048) [2048,3072) [3072,4096)
  gemm_wmma<0><<<dim3(2048 / 64, BL / 128), blk, 0, stream>>>(x, D_DIM, wq, NH * HD,  qkv,        4096, D_DIM);
  gemm_wmma<0><<<dim3(1024 / 64, BL / 128), blk, 0, stream>>>(x, D_DIM, wk, KHD * HD, qkv + 2048, 4096, D_DIM);
  gemm_wmma<0><<<dim3(1024 / 64, BL / 128), blk, 0, stream>>>(x, D_DIM, wv, KHD * HD, qkv + 3072, 4096, D_DIM);
  // RMS-norm + RoPE (Q,K); V: bf16 convert + transpose
  normrope_kernel<<<BL * NH  / 8, blk, 0, stream>>>(qkv, 0,    NH,  qnw, pos, qb, 0);
  normrope_kernel<<<BL * KHD / 8, blk, 0, stream>>>(qkv, 2048, KHD, knw, pos, kb, 1);
  normrope_kernel<<<BL * KHD / 8, blk, 0, stream>>>(qkv, 3072, KHD, knw, pos, vb, 2);
  // flash attention (TDM-staged K/V tiles)
  flash_attn<<<dim3(L_DIM / 64, NH, B_DIM), dim3(128), 0, stream>>>(qb, kb, vb, ao);
  // output projection: AO(4096x2048 bf16, TDM-staged) * Wo(2048x2048) -> out
  gemm_wmma<1><<<dim3(D_DIM / 64, BL / 128), blk, 0, stream>>>(ao, NH * HD, wo, D_DIM, out, D_DIM, NH * HD);
}
